// SDgraphConv_71459665871199
// MI455X (gfx1250) — matrix operand order repaired and checked by reading, not compiled, not run on hardware
//
#include <hip/hip_runtime.h>

// ---------------------------------------------------------------------------
// SDgraphConv for MI455X (gfx1250): all GEMM-shaped work (MLPs + kNN Gram)
// routed through v_wmma_f32_16x16x32_bf16. wave32, 16x16x32 tiles.
// Chunk sizes (EC=64K edges, PANEL=512 kNN cols) keep inter-kernel
// activations resident in the 192MB global L2.
// ---------------------------------------------------------------------------

typedef __bf16 bf16_t;
typedef __attribute__((ext_vector_type(8)))  __bf16 v8bf;
typedef __attribute__((ext_vector_type(16))) __bf16 v16bf;
typedef __attribute__((ext_vector_type(8)))  float  v8f;

#define N_NODES 16384
#define N_EDGES 262144
#define DIM     128
#define DE_DIM  64
#define H_DIM   256
#define K_NN    16
#define EC      65536      // edge chunk (rows per GEMM pass)
#define PANEL   512        // kNN column panel

// A fragment (16x32 bf16): lane holds K [8h, +8) and [16+8h, +8) of its row.
__device__ __forceinline__ v16bf load_a_frag(const bf16_t* __restrict__ ap) {
  v8bf lo = *(const v8bf*)(ap);
  v8bf hi = *(const v8bf*)(ap + 16);
  return __builtin_shufflevector(lo, hi, 0, 1, 2, 3, 4, 5, 6, 7,
                                 8, 9, 10, 11, 12, 13, 14, 15);
}

// ---------------------------------------------------------------------------
// Generic WMMA GEMM:  C = act(A[MxK] @ Bt^T + bias)
//   A : bf16 row-major MxK (compact, lda=K)
//   Bt: bf16 row-major NcxK (W pre-transposed; a B fragment lane reads
//       two runs of 8 contiguous bf16 = two 16B loads)
//   C : OutT row-major MxNc
// Block = 256 threads = 8 waves; block tile 256(M) x 64(N).
// Wave w owns a 32x64 strip: 2 A-frags x 4 B-frags = 8 WMMAs per K-step
// (12 b128 loads per 8 WMMAs).
// ---------------------------------------------------------------------------
template <typename OutT, bool RELU>
__global__ __launch_bounds__(256) void wmma_gemm(
    const bf16_t* __restrict__ A, const bf16_t* __restrict__ Bt,
    const float* __restrict__ bias, OutT* __restrict__ C,
    int M, int K, int Nc) {
  const int lane = threadIdx.x & 31;
  const int wave = threadIdx.x >> 5;
  const int half = lane >> 4;       // lane group (0: lanes 0-15, 1: 16-31)
  const int l16  = lane & 15;
  const int m0 = blockIdx.x * 256 + wave * 32;
  const int n0 = blockIdx.y * 64;

  const v8f zero = {0.f, 0.f, 0.f, 0.f, 0.f, 0.f, 0.f, 0.f};
  v8f acc[2][4] = {{zero, zero, zero, zero}, {zero, zero, zero, zero}};

  // hoisted base pointers (K-invariant part of the addressing)
  const bf16_t* arow0 = A + (size_t)(m0 + l16) * K + 8 * half;        // rows m0..+16
  const bf16_t* arow1 = A + (size_t)(m0 + 16 + l16) * K + 8 * half;   // rows +16..+32
  const bf16_t* brow  = Bt + (size_t)(n0 + l16) * K + 8 * half;       // col n0+l16

  for (int k0 = 0; k0 < K; k0 += 32) {
    const v16bf a0 = load_a_frag(arow0 + k0);
    const v16bf a1 = load_a_frag(arow1 + k0);
#pragma unroll
    for (int t = 0; t < 4; ++t) {
      // B fragment (32x16): lane = column n0+t*16+l16, K [k0+8h, +8) and [k0+16+8h, +8)
      const v16bf b = load_a_frag(brow + (size_t)(t * 16) * K + k0);
      acc[0][t] = __builtin_amdgcn_wmma_f32_16x16x32_bf16(
          false, a0, false, b, (short)0, acc[0][t], false, false);
      acc[1][t] = __builtin_amdgcn_wmma_f32_16x16x32_bf16(
          false, a1, false, b, (short)0, acc[1][t], false, false);
    }
  }

#pragma unroll
  for (int t = 0; t < 4; ++t) {
    const int n = n0 + t * 16 + l16;
    const float bv = bias ? bias[n] : 0.0f;
#pragma unroll
    for (int r = 0; r < 2; ++r) {
#pragma unroll
      for (int v = 0; v < 8; ++v) {
        const int m = m0 + r * 16 + v + 8 * half;  // C layout: VGPR v -> row v+8h
        float val = acc[r][t][v] + bv;
        if (RELU) val = fmaxf(val, 0.0f);
        C[(size_t)m * Nc + n] = (OutT)val;
      }
    }
  }
}

// ---------------------------------------------------------------------------
// Small helper kernels
// ---------------------------------------------------------------------------
__global__ void k_f32_to_bf16(const float* __restrict__ s,
                              bf16_t* __restrict__ d, int n) {
  int i = blockIdx.x * blockDim.x + threadIdx.x;
  if (i < n) d[i] = (bf16_t)s[i];
}

// W[KxNc] f32 row-major -> Wt[NcxK] bf16 row-major
__global__ void k_wtrans(const float* __restrict__ W, bf16_t* __restrict__ Wt,
                         int K, int Nc) {
  int i = blockIdx.x * blockDim.x + threadIdx.x;
  if (i < K * Nc) {
    int k = i / Nc, n = i - k * Nc;
    Wt[(size_t)n * K + k] = (bf16_t)W[i];
  }
}

__global__ void k_zero_f32(float* __restrict__ p, int n) {
  int i = blockIdx.x * blockDim.x + threadIdx.x;
  if (i < n) p[i] = 0.0f;
}

// msg_in[e] = concat(ea[e] (64), xs[src]-xs[dst] (128)) -> bf16, ld=192
__global__ void k_static_gather(const bf16_t* __restrict__ ea,
                                const bf16_t* __restrict__ xs,
                                const long long* __restrict__ srcI,
                                const long long* __restrict__ dstI,
                                bf16_t* __restrict__ out, int e0) {
  const int e = e0 + blockIdx.x;
  const int c = threadIdx.x;  // 0..191
  bf16_t v;
  if (c < DE_DIM) {
    v = ea[(size_t)e * DE_DIM + c];
  } else {
    const int c2 = c - DE_DIM;
    const long long s = srcI[e], d = dstI[e];
    v = (bf16_t)((float)xs[(size_t)s * DIM + c2] -
                 (float)xs[(size_t)d * DIM + c2]);
  }
  out[(size_t)blockIdx.x * (DE_DIM + DIM) + c] = v;
}

__global__ void k_scatter_add(const bf16_t* __restrict__ m,
                              const long long* __restrict__ dstI,
                              float* __restrict__ agg, int e0) {
  const int idx = blockIdx.x * blockDim.x + threadIdx.x;  // EC*128 threads
  const int e = idx >> 7, c = idx & 127;
  const long long d = dstI[e0 + e];
  atomicAdd(&agg[(size_t)d * DIM + c], (float)m[idx]);
}

__global__ void k_sqnorm(const bf16_t* __restrict__ x, float* __restrict__ sq) {
  const int r = blockIdx.x * blockDim.x + threadIdx.x;
  if (r >= N_NODES) return;
  float s = 0.0f;
  for (int c = 0; c < DIM; ++c) {
    const float v = (float)x[(size_t)r * DIM + c];
    s += v * v;
  }
  sq[r] = s;
}

// merge one 512-col Gram panel into running top-16 (smallest dist) per row
__global__ void k_topk(const float* __restrict__ gram,
                       const float* __restrict__ sq, float* __restrict__ td,
                       int* __restrict__ ti, int col0, int first) {
  const int r = blockIdx.x * blockDim.x + threadIdx.x;
  if (r >= N_NODES) return;
  float best[K_NN];
  int bid[K_NN];
  if (first) {
    for (int j = 0; j < K_NN; ++j) { best[j] = 3.4e38f; bid[j] = 0; }
  } else {
    for (int j = 0; j < K_NN; ++j) {
      best[j] = td[(size_t)r * K_NN + j];
      bid[j]  = ti[(size_t)r * K_NN + j];
    }
  }
  const float sr = sq[r];
  for (int j = 0; j < PANEL; ++j) {
    const int c = col0 + j;
    const float d = sr - 2.0f * gram[(size_t)r * PANEL + j] + sq[c];
    if (d < best[K_NN - 1]) {
      int p = K_NN - 1;
      while (p > 0 && best[p - 1] > d) {
        best[p] = best[p - 1];
        bid[p]  = bid[p - 1];
        --p;
      }
      best[p] = d;
      bid[p] = c;
    }
  }
  for (int j = 0; j < K_NN; ++j) {
    td[(size_t)r * K_NN + j] = best[j];
    ti[(size_t)r * K_NN + j] = bid[j];
  }
}

// msg_in[t] = concat(xd[r], xd[src]-xd[r]); dyn edge t, r = t/16, src = ti[t]
__global__ void k_dyn_gather(const bf16_t* __restrict__ xd,
                             const int* __restrict__ ti,
                             bf16_t* __restrict__ out, int t0) {
  const int t = t0 + blockIdx.x;
  const int r = t >> 4;
  const int s = ti[t];
  const int c = threadIdx.x;  // 0..255
  float v;
  if (c < DIM) {
    v = (float)xd[(size_t)r * DIM + c];
  } else {
    const int c2 = c - DIM;
    v = (float)xd[(size_t)s * DIM + c2] - (float)xd[(size_t)r * DIM + c2];
  }
  out[(size_t)blockIdx.x * (2 * DIM) + c] = (bf16_t)v;
}

// contiguous segment sum: agg[r] = sum_j m[(r-r0)*16+j]
__global__ void k_dyn_segsum(const bf16_t* __restrict__ m,
                             float* __restrict__ agg, int r0) {
  const int rl = blockIdx.x;
  const int c = threadIdx.x;  // 0..127
  float s = 0.0f;
#pragma unroll
  for (int j = 0; j < K_NN; ++j)
    s += (float)m[((size_t)rl * K_NN + j) * DIM + c];
  agg[(size_t)(r0 + rl) * DIM + c] = s;
}

__global__ void k_concat(const bf16_t* __restrict__ a,
                         const bf16_t* __restrict__ b,
                         bf16_t* __restrict__ out) {
  const int r = blockIdx.x;
  const int c = threadIdx.x;  // 0..255
  out[(size_t)r * 256 + c] =
      (c < DIM) ? a[(size_t)r * DIM + c] : b[(size_t)r * DIM + (c - DIM)];
}

// ---------------------------------------------------------------------------
// Host orchestration
// ---------------------------------------------------------------------------
static inline void gemm_bf16_relu(const bf16_t* A, const bf16_t* Bt,
                                  const float* bias, bf16_t* C, int M, int K,
                                  int Nc, hipStream_t st) {
  dim3 g(M / 256, Nc / 64);
  wmma_gemm<bf16_t, true><<<g, 256, 0, st>>>(A, Bt, bias, C, M, K, Nc);
}

extern "C" void kernel_launch(void* const* d_in, const int* in_sizes, int n_in,
                              void* d_out, int out_size, void* d_ws,
                              size_t ws_size, hipStream_t stream) {
  (void)in_sizes; (void)n_in; (void)out_size; (void)ws_size;

  // ---- inputs (setup_inputs order) ----
  const float* x   = (const float*)d_in[0];
  const float* eat = (const float*)d_in[1];
  const float* sW1 = (const float*)d_in[2];  const float* sb1 = (const float*)d_in[3];
  const float* sW2 = (const float*)d_in[4];  const float* sb2 = (const float*)d_in[5];
  const float* uW1 = (const float*)d_in[6];  const float* ub1 = (const float*)d_in[7];
  const float* uW2 = (const float*)d_in[8];  const float* ub2 = (const float*)d_in[9];
  const float* rW  = (const float*)d_in[10]; const float* rb  = (const float*)d_in[11];
  const float* dW1 = (const float*)d_in[12]; const float* db1 = (const float*)d_in[13];
  const float* dW2 = (const float*)d_in[14]; const float* db2 = (const float*)d_in[15];
  const float* dUW1 = (const float*)d_in[16]; const float* dUb1 = (const float*)d_in[17];
  const float* dUW2 = (const float*)d_in[18]; const float* dUb2 = (const float*)d_in[19];
  const float* fW1 = (const float*)d_in[20]; const float* fb1 = (const float*)d_in[21];
  const float* fW2 = (const float*)d_in[22]; const float* fb2 = (const float*)d_in[23];
  const long long* ei = (const long long*)d_in[24];  // int64 [2, E]
  const long long* srcI = ei;
  const long long* dstI = ei + N_EDGES;

  // ---- workspace carve ----
  char* base = (char*)d_ws;
  size_t off = 0;
  auto carve = [&](size_t bytes) -> void* {
    void* p = base + off;
    off += (bytes + 255) & ~(size_t)255;
    return p;
  };
  bf16_t* sW1t[3]; bf16_t* sW2t[3]; bf16_t* uW1t[3]; bf16_t* uW2t[3];
  bf16_t* rWt[2];  bf16_t* dW1t[2]; bf16_t* dW2t[2]; bf16_t* dUW1t[2]; bf16_t* dUW2t[2];
  for (int i = 0; i < 3; ++i) sW1t[i] = (bf16_t*)carve((size_t)192 * 256 * 2);
  for (int i = 0; i < 3; ++i) sW2t[i] = (bf16_t*)carve((size_t)256 * 128 * 2);
  for (int i = 0; i < 3; ++i) uW1t[i] = (bf16_t*)carve((size_t)128 * 256 * 2);
  for (int i = 0; i < 3; ++i) uW2t[i] = (bf16_t*)carve((size_t)256 * 128 * 2);
  for (int i = 0; i < 2; ++i) rWt[i]  = (bf16_t*)carve((size_t)64 * 64 * 2);
  for (int i = 0; i < 2; ++i) dW1t[i] = (bf16_t*)carve((size_t)256 * 256 * 2);
  for (int i = 0; i < 2; ++i) dW2t[i] = (bf16_t*)carve((size_t)256 * 128 * 2);
  for (int i = 0; i < 2; ++i) dUW1t[i] = (bf16_t*)carve((size_t)128 * 256 * 2);
  for (int i = 0; i < 2; ++i) dUW2t[i] = (bf16_t*)carve((size_t)256 * 128 * 2);
  bf16_t* fW1t = (bf16_t*)carve((size_t)256 * 256 * 2);
  bf16_t* fW2t = (bf16_t*)carve((size_t)256 * 128 * 2);

  bf16_t* xs   = (bf16_t*)carve((size_t)N_NODES * DIM * 2);
  bf16_t* xd   = (bf16_t*)carve((size_t)N_NODES * DIM * 2);
  bf16_t* eaA  = (bf16_t*)carve((size_t)N_EDGES * DE_DIM * 2);
  bf16_t* eaB  = (bf16_t*)carve((size_t)N_EDGES * DE_DIM * 2);
  bf16_t* msgin = (bf16_t*)carve((size_t)EC * 256 * 2);
  bf16_t* hid   = (bf16_t*)carve((size_t)EC * 256 * 2);
  bf16_t* mch   = (bf16_t*)carve((size_t)EC * 128 * 2);
  float*  agg   = (float*)carve((size_t)N_NODES * DIM * 4);
  bf16_t* aggb  = (bf16_t*)carve((size_t)N_NODES * DIM * 2);
  bf16_t* nhid  = (bf16_t*)carve((size_t)N_NODES * 256 * 2);
  float*  sq    = (float*)carve((size_t)N_NODES * 4);
  float*  gram  = (float*)carve((size_t)N_NODES * PANEL * 4);
  float*  td    = (float*)carve((size_t)N_NODES * K_NN * 4);
  int*    ti    = (int*)carve((size_t)N_NODES * K_NN * 4);
  bf16_t* fin   = (bf16_t*)carve((size_t)N_NODES * 256 * 2);

  auto wt = [&](const float* Ws, bf16_t* Wd, int K, int Nc) {
    const int n = K * Nc;
    k_wtrans<<<(n + 255) / 256, 256, 0, stream>>>(Ws, Wd, K, Nc);
  };

  // ---- weight prep (f32 -> bf16, transposed) ----
  for (int i = 0; i < 3; ++i) {
    wt(sW1 + (size_t)i * 192 * 256, sW1t[i], 192, 256);
    wt(sW2 + (size_t)i * 256 * 128, sW2t[i], 256, 128);
    wt(uW1 + (size_t)i * 128 * 256, uW1t[i], 128, 256);
    wt(uW2 + (size_t)i * 256 * 128, uW2t[i], 256, 128);
  }
  for (int i = 0; i < 2; ++i) {
    wt(rW + (size_t)i * 64 * 64, rWt[i], 64, 64);
    wt(dW1 + (size_t)i * 256 * 256, dW1t[i], 256, 256);
    wt(dW2 + (size_t)i * 256 * 128, dW2t[i], 256, 128);
    wt(dUW1 + (size_t)i * 128 * 256, dUW1t[i], 128, 256);
    wt(dUW2 + (size_t)i * 256 * 128, dUW2t[i], 256, 128);
  }
  wt(fW1, fW1t, 256, 256);
  wt(fW2, fW2t, 256, 128);

  // ---- activation prep ----
  k_f32_to_bf16<<<(N_NODES * DIM) / 256, 256, 0, stream>>>(x, xs, N_NODES * DIM);
  k_f32_to_bf16<<<(N_NODES * DIM) / 256, 256, 0, stream>>>(x, xd, N_NODES * DIM);
  k_f32_to_bf16<<<(N_EDGES * DE_DIM) / 256, 256, 0, stream>>>(eat, eaA,
                                                              N_EDGES * DE_DIM);

  // ---- static branch: 3 StaticConv + 2 edge refiners ----
  bf16_t* ea_cur = eaA;
  bf16_t* ea_nxt = eaB;
  for (int i = 0; i < 3; ++i) {
    k_zero_f32<<<(N_NODES * DIM) / 256, 256, 0, stream>>>(agg, N_NODES * DIM);
    for (int ch = 0; ch < N_EDGES / EC; ++ch) {
      const int e0 = ch * EC;
      k_static_gather<<<EC, 192, 0, stream>>>(ea_cur, xs, srcI, dstI, msgin, e0);
      gemm_bf16_relu(msgin, sW1t[i], sb1 + (size_t)i * 256, hid, EC, 192, 256, stream);
      gemm_bf16_relu(hid, sW2t[i], sb2 + (size_t)i * 128, mch, EC, 256, 128, stream);
      k_scatter_add<<<(EC * 128) / 256, 256, 0, stream>>>(mch, dstI, agg, e0);
    }
    k_f32_to_bf16<<<(N_NODES * DIM) / 256, 256, 0, stream>>>(agg, aggb, N_NODES * DIM);
    gemm_bf16_relu(aggb, uW1t[i], ub1 + (size_t)i * 256, nhid, N_NODES, 128, 256, stream);
    gemm_bf16_relu(nhid, uW2t[i], ub2 + (size_t)i * 128, xs, N_NODES, 256, 128, stream);
    if (i < 2) {
      gemm_bf16_relu(ea_cur, rWt[i], rb + (size_t)i * 64, ea_nxt, N_EDGES, 64, 64, stream);
      bf16_t* t = ea_cur; ea_cur = ea_nxt; ea_nxt = t;
    }
  }

  // ---- dynamic branch: 2 DynamicConv with WMMA-based kNN ----
  for (int i = 0; i < 2; ++i) {
    k_sqnorm<<<(N_NODES + 255) / 256, 256, 0, stream>>>(xd, sq);
    for (int p = 0; p < N_NODES / PANEL; ++p) {
      // Gram panel: xd (NxD) @ (xd panel rows)^T -> N x PANEL, f32, no act.
      // 32MB panel stays resident in the 192MB L2 for the top-k pass.
      dim3 g(N_NODES / 256, PANEL / 64);
      wmma_gemm<float, false><<<g, 256, 0, stream>>>(
          xd, xd + (size_t)p * PANEL * DIM, nullptr, gram, N_NODES, DIM, PANEL);
      k_topk<<<(N_NODES + 255) / 256, 256, 0, stream>>>(gram, sq, td, ti,
                                                        p * PANEL, p == 0);
    }
    for (int ch = 0; ch < (N_NODES * K_NN) / EC; ++ch) {
      const int t0 = ch * EC;
      const int r0 = t0 / K_NN;
      k_dyn_gather<<<EC, 256, 0, stream>>>(xd, ti, msgin, t0);
      gemm_bf16_relu(msgin, dW1t[i], db1 + (size_t)i * 256, hid, EC, 256, 256, stream);
      gemm_bf16_relu(hid, dW2t[i], db2 + (size_t)i * 128, mch, EC, 256, 128, stream);
      k_dyn_segsum<<<EC / K_NN, 128, 0, stream>>>(mch, agg, r0);
    }
    k_f32_to_bf16<<<(N_NODES * DIM) / 256, 256, 0, stream>>>(agg, aggb, N_NODES * DIM);
    gemm_bf16_relu(aggb, dUW1t[i], dUb1 + (size_t)i * 256, nhid, N_NODES, 128, 256, stream);
    gemm_bf16_relu(nhid, dUW2t[i], dUb2 + (size_t)i * 128, xd, N_NODES, 256, 128, stream);
  }

  // ---- fuse: concat(xs, xd) -> MLP -> d_out (f32 N x 128) ----
  k_concat<<<N_NODES, 256, 0, stream>>>(xs, xd, fin);
  gemm_bf16_relu(fin, fW1t, fb1, nhid, N_NODES, 256, 256, stream);
  {
    dim3 g(N_NODES / 256, 128 / 64);
    wmma_gemm<float, true><<<g, 256, 0, stream>>>(nhid, fW2t, fb2,
                                                  (float*)d_out, N_NODES, 256, 128);
  }
}
